// RelationHead_11570641896187
// MI455X (gfx1250) — matrix operand order repaired and checked by reading, not compiled, not run on hardware
//
#include <hip/hip_runtime.h>
#include <hip/hip_bf16.h>
#include <math.h>

typedef __attribute__((ext_vector_type(16))) _Float16 v16h;
typedef __attribute__((ext_vector_type(8)))  _Float16 v8h;
typedef __attribute__((ext_vector_type(8)))  float    v8f;

#define NQ   128          // number of queries N
#define HH   192
#define WW   192
#define PIX  (HH*WW)      // 36864 pixels
#define CLS  128
#define POSD 64
#define PB   128          // pixels per block in main kernel
#define LDS_STRIDE 136    // halves; 136*2=272B => 16B-aligned rows

// Fast sigmoid: v_mul + v_exp + v_add + v_rcp (no IEEE div expansion).
// rcp is ~1ulp; B matrix is f16-quantized anyway, so this is precision-free.
__device__ __forceinline__ float sigm(float x) {
    return __builtin_amdgcn_rcpf(1.0f + __expf(-x));
}

// ---------------------------------------------------------------------------
// K1: overlap[i][j] = sigmoid( cls_dot(i,j) + pos_mlp(i,j) )
// grid = 128 blocks (one per i), 128 threads (one per j)
// ---------------------------------------------------------------------------
__global__ __launch_bounds__(128) void k_overlap(
        const float* __restrict__ bbox,
        const float* __restrict__ WU, const float* __restrict__ WV,
        const float* __restrict__ Wpos, const float* __restrict__ WP,
        const int*   __restrict__ cls,
        float* __restrict__ ov)
{
    __shared__ float u[CLS];        // relu(W_U[ci]) * W_P[0:128]
    __shared__ float wpos[4*POSD];  // W_pos, 4x64
    __shared__ float wp2[POSD];     // W_P[128:192]

    const int i = blockIdx.x;
    const int t = threadIdx.x;
    const int ci = cls[i] - 1;

    u[t] = fmaxf(WU[ci*CLS + t], 0.0f) * WP[t];
    wpos[t]       = Wpos[t];
    wpos[t + 128] = Wpos[t + 128];
    if (t < POSD) wp2[t] = WP[CLS + t];
    __syncthreads();

    const int j  = t;
    const int cj = cls[j] - 1;

    float s1 = 0.0f;
    #pragma unroll 8
    for (int c = 0; c < CLS; ++c)
        s1 += u[c] * fmaxf(WV[cj*CLS + c], 0.0f);

    const float xi0 = bbox[4*i+0], yi0 = bbox[4*i+1], xi1 = bbox[4*i+2], yi1 = bbox[4*i+3];
    const float xj0 = bbox[4*j+0], yj0 = bbox[4*j+1], xj1 = bbox[4*j+2], yj1 = bbox[4*j+3];
    const float wi = xi1 - xi0, hi = yi1 - yi0;
    const float wj = xj1 - xj0, hj = yj1 - yj0;
    const float dx = -(((xi0 + xi1) - (xj0 + xj1)) * 0.5f) * __builtin_amdgcn_rcpf(wi);
    const float dy = -(((yi0 + yi1) - (yj0 + yj1)) * 0.5f) * __builtin_amdgcn_rcpf(hi);
    const float dw = __logf(wj / wi);
    const float dh = __logf(hj / hi);

    float s2 = 0.0f;
    #pragma unroll 4
    for (int d = 0; d < POSD; ++d) {
        float v = dx*wpos[d] + dy*wpos[POSD + d] + dw*wpos[2*POSD + d] + dh*wpos[3*POSD + d];
        s2 += fmaxf(v, 0.0f) * wp2[d];
    }

    ov[i*NQ + j] = sigm(s1 + s2);
}

// ---------------------------------------------------------------------------
// K2: rel = relu(ov - ov^T), f16, pre-swizzled into WMMA A-matrix lane layout.
// A 16x32 f16 layout (ISA 7.12.2): lanes 0-15 -> M=0..15, lanes 16-31 same M,
// K offset +8; VGPR v<4: K=2v+p (+base), v>=4: K=16+2(v-4)+p (+base).
// wsA index: ((mtile*4 + ktile)*32 + lane)*16 + half   (32B contiguous / lane)
// ---------------------------------------------------------------------------
__global__ __launch_bounds__(256) void k_frag(
        const float* __restrict__ ov, _Float16* __restrict__ wsA)
{
    const int t    = blockIdx.x*256 + threadIdx.x;  // 0..1023
    const int mt   = t >> 7;                        // M tile 0..7
    const int kt   = (t >> 5) & 3;                  // K tile 0..3
    const int lane = t & 31;
    const int row  = mt*16 + (lane & 15);
    const int koff = kt*32 + ((lane & 16) ? 8 : 0);

    _Float16* dst = wsA + ((mt*4 + kt)*32 + lane)*16;
    #pragma unroll
    for (int h = 0; h < 16; ++h) {
        const int v = h >> 1, p = h & 1;
        const int kl = (v < 4) ? (2*v + p) : (16 + 2*(v - 4) + p);
        const int k  = koff + kl;
        const float r = fmaxf(ov[row*NQ + k] - ov[k*NQ + row], 0.0f);
        dst[h] = (_Float16)r;
    }
}

// ---------------------------------------------------------------------------
// K3: fused  out = x - x*sigmoid(x)*(rel @ sigmoid(x))  over one pass of x.
// 288 blocks x 256 threads (8 waves). Block owns 128 contiguous pixels.
// Wave w owns output rows 16w..16w+15 (one A M-tile, K=128 => 4 WMMAs/strip).
// ---------------------------------------------------------------------------
__global__ __launch_bounds__(256) void k_main(
        const float*   __restrict__ logits,
        const _Float16* __restrict__ wsA,
        float* __restrict__ out)
{
    __shared__ __attribute__((aligned(32))) _Float16 ldsB[PB * LDS_STRIDE]; // [p][m]

    const int pb  = blockIdx.x * PB;
    const int tid = threadIdx.x;

    // Phase 1: coalesced fp32 loads of the 128(m) x 128(p) logits tile,
    // sigmoid -> f16, stored transposed [p][m] for contiguous B fragments.
    #pragma unroll 4
    for (int e = tid; e < NQ * PB; e += 256) {
        const int m = e >> 7;          // K row
        const int p = e & (PB - 1);    // pixel within tile
        const float x = logits[m*PIX + pb + p];
        ldsB[p*LDS_STRIDE + m] = (_Float16)sigm(x);
    }
    __syncthreads();

    const int w     = tid >> 5;
    const int lane  = tid & 31;
    const int rbase = w * 16;

    // A fragments: 4 K-tiles, 32B contiguous per lane, 32B-aligned.
    v16h a[4];
    #pragma unroll
    for (int kt = 0; kt < 4; ++kt)
        a[kt] = *(const v16h*)(wsA + ((w*4 + kt)*32 + lane)*16);

    const int pcol  = lane & 15;
    const int mhalf = (lane & 16) ? 8 : 0;    // B: lanes 16-31 hold K+8
    const int rhi   = (lane & 16) ? 8  : 0;   // D: lanes 16-31 hold M+8

    for (int s = 0; s < PB/16; ++s) {
        const int p   = s*16 + pcol;
        const int col = pb + s*16 + pcol;

        // Hoist epilogue loads: in flight while DS + WMMA issue.
        float xg[8];
        #pragma unroll
        for (int g = 0; g < 8; ++g)
            xg[g] = logits[(rbase + g + rhi)*PIX + col];

        v8f acc = {};
        #pragma unroll
        for (int kt = 0; kt < 4; ++kt) {
            // B 32x16 f16 layout: lane holds col N=lane%16, K = base..base+7
            // and base+16..base+23 in LDS (transposed store) -> two aligned b128 loads.
            const _Float16* src = &ldsB[p*LDS_STRIDE + kt*32 + mhalf];
            union { v16h v; v8h h[2]; } b;
            b.h[0] = *(const v8h*)(src);
            b.h[1] = *(const v8h*)(src + 16);
            acc = __builtin_amdgcn_wmma_f32_16x16x32_f16(
                    /*neg_a=*/false, a[kt], /*neg_b=*/false, b.v,
                    /*c_mod=*/(short)0, acc, /*reuse_a=*/false, /*reuse_b=*/false);
        }

        // Epilogue: C/D layout VGPR g -> M = rbase+g(+8), N = lane%16.
        #pragma unroll
        for (int g = 0; g < 8; ++g) {
            const float x = xg[g];
            const float t = x * sigm(x);
            out[(rbase + g + rhi)*PIX + col] = fmaf(-t, acc[g], x);
        }
    }
}

// ---------------------------------------------------------------------------
extern "C" void kernel_launch(void* const* d_in, const int* in_sizes, int n_in,
                              void* d_out, int out_size, void* d_ws, size_t ws_size,
                              hipStream_t stream)
{
    const float* logits = (const float*)d_in[0];   // (1,128,192,192) f32
    const float* bbox   = (const float*)d_in[1];   // (128,4) f32
    const float* WU     = (const float*)d_in[2];   // (80,128) f32
    const float* WV     = (const float*)d_in[3];   // (80,128) f32
    const float* Wpos   = (const float*)d_in[4];   // (4,64) f32
    const float* WP     = (const float*)d_in[5];   // (192,1) f32
    const int*   cls    = (const int*)d_in[6];     // (128,) i32
    float* out = (float*)d_out;

    float*    ov  = (float*)d_ws;                                   // 64 KB
    _Float16* wsA = (_Float16*)((char*)d_ws + NQ*NQ*sizeof(float)); // 32 KB

    k_overlap<<<NQ, 128, 0, stream>>>(bbox, WU, WV, Wpos, WP, cls, ov);
    k_frag<<<4, 256, 0, stream>>>(ov, wsA);
    k_main<<<PIX / PB, 256, 0, stream>>>(logits, wsA, out);
    (void)in_sizes; (void)n_in; (void)out_size; (void)ws_size;
}
